// WiKG_40123584479601
// MI455X (gfx1250) — compile-verified, matrix operand order checked by reading, not run on hardware
//
#include <hip/hip_runtime.h>
#include <math.h>
#include <stdint.h>

// ---------------- types ----------------
typedef _Float16 f16_t;
typedef __attribute__((ext_vector_type(16))) _Float16 v16h;
typedef __attribute__((ext_vector_type(8)))  _Float16 v8h;
typedef __attribute__((ext_vector_type(8)))  float    v8f;
typedef int i32x4_vs __attribute__((vector_size(16)));   // matches builtin's pointee type

#define CAT16(lo, hi) __builtin_shufflevector((lo), (hi), 0,1,2,3,4,5,6,7,8,9,10,11,12,13,14,15)

#define NROWS 16384
#define DIN   384
#define DDIM  512
#define TOPK  6
#define ATT_SCALE 0.04419417382415922f /* 512^-0.5 */

// ---------------- CDNA5 async global->LDS path ----------------
#ifdef __has_builtin
#if __has_builtin(__builtin_amdgcn_global_load_async_to_lds_b128)
#define HAVE_ASYNC_LDS 1
#endif
#endif
#ifndef HAVE_ASYNC_LDS
#define HAVE_ASYNC_LDS 0
#endif

#define ASG __attribute__((address_space(1)))
#define ASL __attribute__((address_space(3)))

__device__ __forceinline__ void async_b128(const f16_t* g, f16_t* l) {
#if HAVE_ASYNC_LDS
    // flat->AS1: same address value; flat->AS3: low 32 bits are the LDS offset
    __builtin_amdgcn_global_load_async_to_lds_b128(
        (ASG i32x4_vs*)(uintptr_t)g,
        (ASL i32x4_vs*)(uint32_t)(uintptr_t)l, 0, 0);
#else
    *(v8h*)l = *(const v8h*)g;
#endif
}

__device__ __forceinline__ void wait_async0() {
#if HAVE_ASYNC_LDS
#ifdef __has_builtin
#if __has_builtin(__builtin_amdgcn_s_wait_asynccnt)
    __builtin_amdgcn_s_wait_asynccnt(0);
#else
    asm volatile("s_wait_asynccnt 0x0" ::: "memory");
#endif
#else
    asm volatile("s_wait_asynccnt 0x0" ::: "memory");
#endif
#endif
}

// ---------------- elementwise helpers ----------------
__global__ void f2h_kernel(const float* __restrict__ in, f16_t* __restrict__ out,
                           int n, float scale) {
    int i = blockIdx.x * blockDim.x + threadIdx.x;
    if (i < n) out[i] = (f16_t)(in[i] * scale);
}

// out[C][R] = (f16) in[R][C]
__global__ void transpose_h_kernel(const float* __restrict__ in, f16_t* __restrict__ out,
                                   int R, int C) {
    int i = blockIdx.x * blockDim.x + threadIdx.x;
    if (i < R * C) {
        int r = i / C, c = i % C;
        out[(size_t)c * R + r] = (f16_t)in[i];
    }
}

__global__ void colsum_kernel(const float* __restrict__ h, float* __restrict__ colsum) {
    int c  = blockIdx.x * 256 + threadIdx.x;   // 0..511
    int r0 = blockIdx.y * 256;
    float acc = 0.f;
    for (int r = 0; r < 256; ++r) acc += h[(size_t)(r0 + r) * DDIM + c];
    atomicAdd(&colsum[c], acc);
}

__global__ void hupdate_kernel(float* __restrict__ h, f16_t* __restrict__ h_hf,
                               const float* __restrict__ colsum) {
    int i = blockIdx.x * blockDim.x + threadIdx.x;
    if (i < NROWS * DDIM) {
        float m = colsum[i & (DDIM - 1)] * (1.0f / (float)NROWS);
        float v = (h[i] + m) * 0.5f;
        h[i] = v;
        h_hf[i] = (f16_t)v;
    }
}

// ---------------- generic WMMA GEMM (ping-pong pipelined, K step 64) ----------------
// out[M,Nc] = epilogue( A[M,Kc](f16) @ Wt[Nc,Kc]^T(f16) + bias )
// wave computes a 16x64 tile; Kc multiple of 64, Nc multiple of 64, M multiple of 16.
__global__ __launch_bounds__(256) void gemm_hf_kernel(
    const f16_t* __restrict__ A, const f16_t* __restrict__ Wt,
    const float* __restrict__ bias, float* __restrict__ out,
    f16_t* __restrict__ out_hf, float hf_scale,
    int M, int Kc, int Nc, int act, int addin)
{
    const int wvg  = blockIdx.x * 8 + (threadIdx.x >> 5);
    const int lane = threadIdx.x & 31;
    const int m    = lane & 15;
    const int hf   = lane >> 4;
    const int colTiles = Nc >> 6;
    const int rowBase  = (wvg / colTiles) << 4;
    const int colBase  = (wvg % colTiles) << 6;

    const f16_t* ap  = A  + (size_t)(rowBase + m) * Kc + 8 * hf;
    const f16_t* wp0 = Wt + (size_t)(colBase + m) * Kc + 8 * hf;

    const v8f zero8 = {0.f,0.f,0.f,0.f,0.f,0.f,0.f,0.f};
    v8f acc[4];
#pragma unroll
    for (int t = 0; t < 4; ++t) acc[t] = zero8;

    // two pipeline stages: S0 covers k in [kk, kk+32), S1 covers [kk+32, kk+64)
    v16h A0, A1, B0[4], B1[4];
    A0 = CAT16(*(const v8h*)(ap), *(const v8h*)(ap + 16));
#pragma unroll
    for (int t = 0; t < 4; ++t) {
        const f16_t* wp = wp0 + (size_t)t * 16 * Kc;
        B0[t] = CAT16(*(const v8h*)(wp), *(const v8h*)(wp + 16));
    }
    A1 = CAT16(*(const v8h*)(ap + 32), *(const v8h*)(ap + 48));
#pragma unroll
    for (int t = 0; t < 4; ++t) {
        const f16_t* wp = wp0 + (size_t)t * 16 * Kc;
        B1[t] = CAT16(*(const v8h*)(wp + 32), *(const v8h*)(wp + 48));
    }

    int kk = 0;
    for (; kk + 64 < Kc; kk += 64) {
        // consume stage0, refill stage0 from kk+64
#pragma unroll
        for (int t = 0; t < 4; ++t)
            acc[t] = __builtin_amdgcn_wmma_f32_16x16x32_f16(
                false, A0, false, B0[t], (short)0, acc[t], false, false);
        A0 = CAT16(*(const v8h*)(ap + kk + 64), *(const v8h*)(ap + kk + 80));
#pragma unroll
        for (int t = 0; t < 4; ++t) {
            const f16_t* wp = wp0 + (size_t)t * 16 * Kc;
            B0[t] = CAT16(*(const v8h*)(wp + kk + 64), *(const v8h*)(wp + kk + 80));
        }
        // consume stage1, refill stage1 from kk+96
#pragma unroll
        for (int t = 0; t < 4; ++t)
            acc[t] = __builtin_amdgcn_wmma_f32_16x16x32_f16(
                false, A1, false, B1[t], (short)0, acc[t], false, false);
        A1 = CAT16(*(const v8h*)(ap + kk + 96), *(const v8h*)(ap + kk + 112));
#pragma unroll
        for (int t = 0; t < 4; ++t) {
            const f16_t* wp = wp0 + (size_t)t * 16 * Kc;
            B1[t] = CAT16(*(const v8h*)(wp + kk + 96), *(const v8h*)(wp + kk + 112));
        }
    }
    // tail: consume the two staged fragment sets
#pragma unroll
    for (int t = 0; t < 4; ++t)
        acc[t] = __builtin_amdgcn_wmma_f32_16x16x32_f16(
            false, A0, false, B0[t], (short)0, acc[t], false, false);
#pragma unroll
    for (int t = 0; t < 4; ++t)
        acc[t] = __builtin_amdgcn_wmma_f32_16x16x32_f16(
            false, A1, false, B1[t], (short)0, acc[t], false, false);

#pragma unroll
    for (int t = 0; t < 4; ++t) {
        int c = colBase + t * 16 + m;
        float bv = bias[c];
#pragma unroll
        for (int r = 0; r < 8; ++r) {
            int rr = rowBase + r + 8 * hf;
            float v = acc[t][r] + bv;
            if (act) v = v > 0.f ? v : 0.01f * v;
            size_t o = (size_t)rr * Nc + c;
            if (addin) v += out[o];
            out[o] = v;
            if (out_hf) out_hf[o] = (f16_t)(v * hf_scale);
        }
    }
}

// ---------------- fused attention GEMM + top-K ----------------
// logits[i,j] = dot(ehs[i,:], et[j,:]) ; per-row top-6 kept in registers.
// Block = 128 threads (4 waves), each wave owns 16 rows; 64 rows / block.
// B tiles (16 columns x K=512) are double-buffered in LDS via async copies.
__global__ __launch_bounds__(128) void attn_topk_kernel(
    const f16_t* __restrict__ ehs,   // [N,512] f16, pre-scaled by ATT_SCALE
    const f16_t* __restrict__ et,    // [N,512] f16
    float* __restrict__ topv, int* __restrict__ topi)
{
    __shared__ __attribute__((aligned(16))) f16_t Bsh[2][16 * 520]; // 2 x (16 cols x 512 K, padded)
    __shared__ float Lsh[4][16][17];                                // per-wave logit tile

    const int tid  = threadIdx.x;
    const int wv   = tid >> 5;
    const int lane = tid & 31;
    const int m    = lane & 15;
    const int hf   = lane >> 4;
    const int rowBase = blockIdx.x * 64 + wv * 16;

    // resident A fragments: this wave's 16 rows, all K=512 (128 VGPRs)
    v16h afrag[16];
    {
        const f16_t* ar = ehs + (size_t)(rowBase + m) * DDIM + 8 * hf;
#pragma unroll
        for (int kk = 0; kk < 16; ++kk) {
            v8h lo = *(const v8h*)(ar + kk * 32);
            v8h hi = *(const v8h*)(ar + kk * 32 + 16);
            afrag[kk] = CAT16(lo, hi);
        }
    }

    float tv[TOPK]; int ti[TOPK];
#pragma unroll
    for (int k = 0; k < TOPK; ++k) { tv[k] = -3.0e38f; ti[k] = 0; }

    const int bcol = tid >> 3;            // 0..15 : which et row (column j)
    const int bseg = (tid & 7) * 64;      // 64-half (128B) segment within the row
    const v8f zero8 = {0.f,0.f,0.f,0.f,0.f,0.f,0.f,0.f};

    // prologue: stage tile j0 = 0 into buffer 0
    {
        const f16_t* src = et + (size_t)bcol * DDIM + bseg;
        f16_t* dst = &Bsh[0][bcol * 520 + bseg];
#pragma unroll
        for (int i = 0; i < 8; ++i) async_b128(src + i * 8, dst + i * 8);
    }

    for (int j0 = 0; j0 < NROWS; j0 += 16) {
        const int cur = (j0 >> 4) & 1;
        wait_async0();                    // this wave's staged copies landed in LDS
        __syncthreads();                  // tile `cur` ready; everyone done with cur^1

        if (j0 + 16 < NROWS) {            // stage next tile while computing this one
            const f16_t* src = et + (size_t)(j0 + 16 + bcol) * DDIM + bseg;
            f16_t* dst = &Bsh[cur ^ 1][bcol * 520 + bseg];
#pragma unroll
            for (int i = 0; i < 8; ++i) async_b128(src + i * 8, dst + i * 8);
        }

        // 16x16 logit tile: 16 K-steps, register double-buffered B fragments
        {
            const f16_t* bb = &Bsh[cur][m * 520 + 8 * hf];
            v16h bcur = CAT16(*(const v8h*)(bb), *(const v8h*)(bb + 16));
            v8f acc = zero8;
#pragma unroll
            for (int kk = 0; kk < 16; ++kk) {
                v16h bthis = bcur;
                if (kk < 15) {
                    const f16_t* bn = bb + (kk + 1) * 32;
                    bcur = CAT16(*(const v8h*)(bn), *(const v8h*)(bn + 16));
                }
                acc = __builtin_amdgcn_wmma_f32_16x16x32_f16(
                    false, afrag[kk], false, bthis, (short)0, acc, false, false);
            }
#pragma unroll
            for (int r = 0; r < 8; ++r)
                Lsh[wv][r + 8 * hf][m] = acc[r];
        }
        __syncthreads();

        if (lane < 16) {                  // lane = local row; scan 16 new logits
#pragma unroll
            for (int j = 0; j < 16; ++j) {
                float v = Lsh[wv][lane][j];
                if (v > tv[TOPK - 1]) {
                    int idx = j0 + j;
#pragma unroll
                    for (int k = 0; k < TOPK; ++k) {
                        if (v > tv[k]) {
                            float tf = tv[k]; tv[k] = v; v = tf;
                            int   tj = ti[k]; ti[k] = idx; idx = tj;
                        }
                    }
                }
            }
        }
    }

    if (lane < 16) {
        int row = rowBase + lane;
#pragma unroll
        for (int k = 0; k < TOPK; ++k) {
            topv[row * TOPK + k] = tv[k];
            topi[row * TOPK + k] = ti[k];
        }
    }
}

// ---------------- neighbor gather / gate / aggregate ----------------
// wave per row; emits fp16 inputs of lin1 (e_h+e_Nh) and lin2 (e_h*e_Nh) directly.
__global__ __launch_bounds__(256) void aggregate_kernel(
    const float* __restrict__ eh, const float* __restrict__ et,
    const float* __restrict__ topv, const int* __restrict__ topi,
    f16_t* __restrict__ sum_hf, f16_t* __restrict__ prod_hf)
{
    const int wv = threadIdx.x >> 5, lane = threadIdx.x & 31;
    const int row = blockIdx.x * 8 + wv;
    const int d0 = lane * 16;

    float ehr[16];
    {
        const float4* q = (const float4*)(eh + (size_t)row * DDIM + d0);
#pragma unroll
        for (int i = 0; i < 4; ++i) {
            float4 v = q[i];
            ehr[i*4+0]=v.x; ehr[i*4+1]=v.y; ehr[i*4+2]=v.z; ehr[i*4+3]=v.w;
        }
    }

    float tvv[TOPK]; int tii[TOPK];
#pragma unroll
    for (int k = 0; k < TOPK; ++k) { tvv[k] = topv[row*TOPK+k]; tii[k] = topi[row*TOPK+k]; }

    float mx = tvv[0];
#pragma unroll
    for (int k = 1; k < TOPK; ++k) mx = fmaxf(mx, tvv[k]);
    float p[TOPK], ps = 0.f;
#pragma unroll
    for (int k = 0; k < TOPK; ++k) { p[k] = expf(tvv[k] - mx); ps += p[k]; }
    float inv = 1.f / ps;
#pragma unroll
    for (int k = 0; k < TOPK; ++k) p[k] *= inv;

    float nb[TOPK][16];
    float ka[TOPK];
#pragma unroll
    for (int k = 0; k < TOPK; ++k) {
        const float4* np = (const float4*)(et + (size_t)tii[k] * DDIM + d0);
        float ln = 0.f, lg = 0.f;
#pragma unroll
        for (int i = 0; i < 4; ++i) {
            float4 v = np[i];
            nb[k][i*4+0]=v.x; nb[k][i*4+1]=v.y; nb[k][i*4+2]=v.z; nb[k][i*4+3]=v.w;
        }
#pragma unroll
        for (int i = 0; i < 16; ++i) {
            float nv = nb[k][i];
            ln += nv;
            float arg = (2.0f - p[k]) * ehr[i] + p[k] * nv; // e_h + eh_r
            lg += tanhf(arg);
        }
        for (int off = 16; off >= 1; off >>= 1) {
            ln += __shfl_xor(ln, off, 32);
            lg += __shfl_xor(lg, off, 32);
        }
        ka[k] = ln * lg;
    }

    float km = ka[0];
#pragma unroll
    for (int k = 1; k < TOPK; ++k) km = fmaxf(km, ka[k]);
    float kp[TOPK], ks = 0.f;
#pragma unroll
    for (int k = 0; k < TOPK; ++k) { kp[k] = expf(ka[k] - km); ks += kp[k]; }
    float kinv = 1.f / ks;
#pragma unroll
    for (int k = 0; k < TOPK; ++k) kp[k] *= kinv;

    f16_t* sp = sum_hf  + (size_t)row * DDIM + d0;
    f16_t* pp = prod_hf + (size_t)row * DDIM + d0;
#pragma unroll
    for (int i = 0; i < 16; ++i) {
        float en = 0.f;
#pragma unroll
        for (int k = 0; k < TOPK; ++k) en += kp[k] * nb[k][i];
        sp[i] = (f16_t)(ehr[i] + en);
        pp[i] = (f16_t)(ehr[i] * en);
    }
}

// ---------------- readout ----------------
__global__ __launch_bounds__(256) void score_kernel(
    const float* __restrict__ r1, const float* __restrict__ w,
    const float* __restrict__ b, float* __restrict__ a)
{
    const int wv = threadIdx.x >> 5, lane = threadIdx.x & 31;
    const int row = blockIdx.x * 8 + wv;
    const float* rp = r1 + (size_t)row * 256 + lane * 8;
    float s = 0.f;
#pragma unroll
    for (int i = 0; i < 8; ++i) s += rp[i] * w[lane * 8 + i];
    for (int off = 16; off >= 1; off >>= 1) s += __shfl_xor(s, off, 32);
    if (lane == 0) a[row] = s + b[0];
}

__global__ __launch_bounds__(1024) void softmax_g_kernel(
    const float* __restrict__ a, float* __restrict__ g, float* __restrict__ pooled)
{
    __shared__ float red[1024];
    const int t = threadIdx.x;
    float mx = -3.0e38f;
    for (int i = t; i < NROWS; i += 1024) mx = fmaxf(mx, a[i]);
    red[t] = mx; __syncthreads();
    for (int s = 512; s > 0; s >>= 1) { if (t < s) red[t] = fmaxf(red[t], red[t + s]); __syncthreads(); }
    mx = red[0]; __syncthreads();
    float sum = 0.f;
    for (int i = t; i < NROWS; i += 1024) sum += expf(a[i] - mx);
    red[t] = sum; __syncthreads();
    for (int s = 512; s > 0; s >>= 1) { if (t < s) red[t] += red[t + s]; __syncthreads(); }
    float inv = 1.0f / red[0];
    for (int i = t; i < NROWS; i += 1024) g[i] = expf(a[i] - mx) * inv;
    if (t < DDIM) pooled[t] = 0.f;
}

__global__ __launch_bounds__(512) void pooled_kernel(
    const float* __restrict__ g, const float* __restrict__ emb, float* __restrict__ pooled)
{
    const int c = threadIdx.x;      // 0..511
    const int r0 = blockIdx.x * 256;
    float acc = 0.f;
    for (int r = 0; r < 256; ++r) acc += g[r0 + r] * emb[(size_t)(r0 + r) * DDIM + c];
    atomicAdd(&pooled[c], acc);
}

__global__ __launch_bounds__(512) void final_kernel(
    const float* __restrict__ pooled, const float* __restrict__ ln_g,
    const float* __restrict__ ln_b, const float* __restrict__ fc_w,
    const float* __restrict__ fc_b, float* __restrict__ out)
{
    __shared__ float red[512];
    const int t = threadIdx.x;
    float v = pooled[t];
    red[t] = v; __syncthreads();
    for (int s = 256; s > 0; s >>= 1) { if (t < s) red[t] += red[t + s]; __syncthreads(); }
    float mu = red[0] * (1.f / 512.f); __syncthreads();
    float d = v - mu;
    red[t] = d * d; __syncthreads();
    for (int s = 256; s > 0; s >>= 1) { if (t < s) red[t] += red[t + s]; __syncthreads(); }
    float var = red[0] * (1.f / 512.f); __syncthreads();
    float nrm = d * rsqrtf(var + 1e-5f) * ln_g[t] + ln_b[t];

    red[t] = nrm * fc_w[t * 2 + 0]; __syncthreads();
    for (int s = 256; s > 0; s >>= 1) { if (t < s) red[t] += red[t + s]; __syncthreads(); }
    float l0 = red[0] + fc_b[0]; __syncthreads();
    red[t] = nrm * fc_w[t * 2 + 1]; __syncthreads();
    for (int s = 256; s > 0; s >>= 1) { if (t < s) red[t] += red[t + s]; __syncthreads(); }
    float l1 = red[0] + fc_b[1];

    if (t == 0) {
        out[0] = l0; out[1] = l1;
        float mm = fmaxf(l0, l1);
        float e0 = expf(l0 - mm), e1 = expf(l1 - mm);
        float iv = 1.f / (e0 + e1);
        out[2] = e0 * iv; out[3] = e1 * iv;
    }
}

// ---------------- host orchestration ----------------
extern "C" void kernel_launch(void* const* d_in, const int* in_sizes, int n_in,
                              void* d_out, int out_size, void* d_ws, size_t ws_size,
                              hipStream_t stream) {
    (void)in_sizes; (void)n_in; (void)out_size; (void)ws_size;
    const float* x      = (const float*)d_in[0];
    const float* fc1_w  = (const float*)d_in[1];
    const float* fc1_b  = (const float*)d_in[2];
    const float* wh_w   = (const float*)d_in[3];
    const float* wh_b   = (const float*)d_in[4];
    const float* wt_w   = (const float*)d_in[5];
    const float* wt_b   = (const float*)d_in[6];
    const float* lin1_w = (const float*)d_in[7];
    const float* lin1_b = (const float*)d_in[8];
    const float* lin2_w = (const float*)d_in[9];
    const float* lin2_b = (const float*)d_in[10];
    const float* att1_w = (const float*)d_in[11];
    const float* att1_b = (const float*)d_in[12];
    const float* att2_w = (const float*)d_in[13];
    const float* att2_b = (const float*)d_in[14];
    const float* ln_g   = (const float*)d_in[15];
    const float* ln_bp  = (const float*)d_in[16];
    const float* fc_w   = (const float*)d_in[17];
    const float* fc_b   = (const float*)d_in[18];

    char* base = (char*)d_ws;
    size_t off = 0;
    auto carve = [&](size_t bytes) -> void* {
        void* p = base + off;
        off += (bytes + 255) & ~(size_t)255;
        return p;
    };

    f16_t* x_hf    = (f16_t*)carve((size_t)NROWS * DIN * 2);
    f16_t* fc1_wt  = (f16_t*)carve((size_t)DDIM * DIN * 2);
    f16_t* wh_wt   = (f16_t*)carve((size_t)DDIM * DDIM * 2);
    f16_t* wt_wt   = (f16_t*)carve((size_t)DDIM * DDIM * 2);
    f16_t* lin1_wt = (f16_t*)carve((size_t)DDIM * DDIM * 2);
    f16_t* lin2_wt = (f16_t*)carve((size_t)DDIM * DDIM * 2);
    f16_t* att1_wt = (f16_t*)carve((size_t)256 * DDIM * 2);
    float* h       = (float*)carve((size_t)NROWS * DDIM * 4);   // reused as emb
    f16_t* h_hf    = (f16_t*)carve((size_t)NROWS * DDIM * 2);   // reused as r1 (fp32 [N,256])
    float* eh      = (float*)carve((size_t)NROWS * DDIM * 4);
    float* et      = (float*)carve((size_t)NROWS * DDIM * 4);
    f16_t* ehs_hf  = (f16_t*)carve((size_t)NROWS * DDIM * 2);   // reused as sum_hf
    f16_t* et_hf   = (f16_t*)carve((size_t)NROWS * DDIM * 2);   // reused as prod_hf
    f16_t* emb_hf  = (f16_t*)carve((size_t)NROWS * DDIM * 2);
    float* topv    = (float*)carve((size_t)NROWS * TOPK * 4);
    int*   topi    = (int*)  carve((size_t)NROWS * TOPK * 4);
    float* av      = (float*)carve((size_t)NROWS * 4);
    float* gv      = (float*)carve((size_t)NROWS * 4);
    float* colsum  = (float*)carve((size_t)DDIM * 4);
    float* pooled  = (float*)carve((size_t)DDIM * 4);

    float* emb     = h;            // h dead after eh/et GEMMs
    f16_t* sum_hf  = ehs_hf;       // dead after attn kernel
    f16_t* prod_hf = et_hf;        // dead after attn kernel
    float* r1      = (float*)h_hf; // N*256 fp32 == N*512 f16 bytes

    // 1) fp16 copies / transposed fp16 weights
    f2h_kernel<<<(NROWS * DIN + 255) / 256, 256, 0, stream>>>(x, x_hf, NROWS * DIN, 1.f);
    transpose_h_kernel<<<(DIN * DDIM + 255) / 256, 256, 0, stream>>>(fc1_w, fc1_wt, DIN, DDIM);
    transpose_h_kernel<<<(DDIM * DDIM + 255) / 256, 256, 0, stream>>>(wh_w, wh_wt, DDIM, DDIM);
    transpose_h_kernel<<<(DDIM * DDIM + 255) / 256, 256, 0, stream>>>(wt_w, wt_wt, DDIM, DDIM);
    transpose_h_kernel<<<(DDIM * DDIM + 255) / 256, 256, 0, stream>>>(lin1_w, lin1_wt, DDIM, DDIM);
    transpose_h_kernel<<<(DDIM * DDIM + 255) / 256, 256, 0, stream>>>(lin2_w, lin2_wt, DDIM, DDIM);
    transpose_h_kernel<<<(DDIM * 256 + 255) / 256, 256, 0, stream>>>(att1_w, att1_wt, DDIM, 256);

    // 2) h = lrelu(x @ fc1_w + b)
    gemm_hf_kernel<<<(NROWS / 16) * (DDIM / 64) / 8, 256, 0, stream>>>(
        x_hf, fc1_wt, fc1_b, h, nullptr, 1.f, NROWS, DIN, DDIM, 1, 0);

    // 3) h = (h + colmean(h)) * 0.5 ; emit fp16 copy
    (void)hipMemsetAsync(colsum, 0, DDIM * 4, stream);
    colsum_kernel<<<dim3(2, 64), 256, 0, stream>>>(h, colsum);
    hupdate_kernel<<<(NROWS * DDIM) / 256, 256, 0, stream>>>(h, h_hf, colsum);

    // 4) e_h, e_t (+ fp16 sidebands; e_h scaled by D^-0.5 for attention)
    gemm_hf_kernel<<<(NROWS / 16) * (DDIM / 64) / 8, 256, 0, stream>>>(
        h_hf, wh_wt, wh_b, eh, ehs_hf, ATT_SCALE, NROWS, DDIM, DDIM, 0, 0);
    gemm_hf_kernel<<<(NROWS / 16) * (DDIM / 64) / 8, 256, 0, stream>>>(
        h_hf, wt_wt, wt_b, et, et_hf, 1.f, NROWS, DDIM, DDIM, 0, 0);

    // 5) fused attention GEMM + top-6 (never materializes N x N)
    attn_topk_kernel<<<NROWS / 64, 128, 0, stream>>>(ehs_hf, et_hf, topv, topi);

    // 6) gather neighbors, gate, aggregate -> fp16 inputs of lin1/lin2
    aggregate_kernel<<<NROWS / 8, 256, 0, stream>>>(eh, et, topv, topi, sum_hf, prod_hf);

    // 7) emb = lrelu(sum@lin1+b) + lrelu(prod@lin2+b)
    gemm_hf_kernel<<<(NROWS / 16) * (DDIM / 64) / 8, 256, 0, stream>>>(
        sum_hf, lin1_wt, lin1_b, emb, nullptr, 1.f, NROWS, DDIM, DDIM, 1, 0);
    gemm_hf_kernel<<<(NROWS / 16) * (DDIM / 64) / 8, 256, 0, stream>>>(
        prod_hf, lin2_wt, lin2_b, emb, emb_hf, 1.f, NROWS, DDIM, DDIM, 1, 1);

    // 8) global-attention readout
    gemm_hf_kernel<<<(NROWS / 16) * (256 / 64) / 8, 256, 0, stream>>>(
        emb_hf, att1_wt, att1_b, r1, nullptr, 1.f, NROWS, DDIM, 256, 1, 0);
    score_kernel<<<NROWS / 8, 256, 0, stream>>>(r1, att2_w, att2_b, av);
    softmax_g_kernel<<<1, 1024, 0, stream>>>(av, gv, pooled);
    pooled_kernel<<<NROWS / 256, 512, 0, stream>>>(gv, emb, pooled);
    final_kernel<<<1, 512, 0, stream>>>(pooled, ln_g, ln_bp, fc_w, fc_b, (float*)d_out);
}
